// Detector_62852551410174
// MI455X (gfx1250) — matrix operand order repaired and checked by reading, not compiled, not run on hardware
//
#include <hip/hip_runtime.h>
#include <hip/hip_bf16.h>
#include <math.h>

// MI455X (gfx1250). Memory-bound: ~100MB k/v streamed once; ~1.6 GFLOP.
// Matrix work: v_wmma_f32_16x16x32_f16. k/q1 staging: global_load_async_to_lds
// (ASYNCcnt). All fragment loads are branchless float4 -> v_cvt_pk_f16_f32.

typedef __attribute__((ext_vector_type(16))) _Float16 v16h;
typedef __attribute__((ext_vector_type(8)))  float    v8f;

constexpr int L_ = 2, N_ = 4, Q_ = 8, F_ = 8, P_ = 257, H_ = 12, C_ = 64;
constexpr int K_ = F_ * P_;       // 2056
constexpr int W_ = H_ * C_;       // 768
constexpr int KPAD = 2080;        // 65 * 32
constexpr int NKB = KPAD / 32;    // 65

__device__ __forceinline__ v8f wmma32(v16h a, v16h b, v8f c) {
  return __builtin_amdgcn_wmma_f32_16x16x32_f16(false, a, false, b, (short)0, c,
                                                false, false);
}

__device__ __forceinline__ float4 ld4(const float* p) {
  return *(const float4*)p;
}

// Async copy of 16B memory->LDS per lane (ASYNCcnt-tracked).
__device__ __forceinline__ void async_b128(const float* gsrc, void* ldst) {
  unsigned d = (unsigned)(size_t)ldst;                 // LDS aperture low bits
  unsigned long long s = (unsigned long long)(size_t)gsrc;
  asm volatile("global_load_async_to_lds_b128 %0, %1, off"
               :: "v"(d), "v"(s) : "memory");
}

// Async copy of 256B (one 64-float row) memory->LDS per lane: the 24-bit
// instruction offset is added to both LDS and global addresses (ISA 15.18).
__device__ __forceinline__ void async_row256(const float* gsrc, void* ldst) {
  unsigned d = (unsigned)(size_t)ldst;
  unsigned long long s = (unsigned long long)(size_t)gsrc;
  asm volatile(
      "global_load_async_to_lds_b128 %0, %1, off\n\t"
      "global_load_async_to_lds_b128 %0, %1, off offset:16\n\t"
      "global_load_async_to_lds_b128 %0, %1, off offset:32\n\t"
      "global_load_async_to_lds_b128 %0, %1, off offset:48\n\t"
      "global_load_async_to_lds_b128 %0, %1, off offset:64\n\t"
      "global_load_async_to_lds_b128 %0, %1, off offset:80\n\t"
      "global_load_async_to_lds_b128 %0, %1, off offset:96\n\t"
      "global_load_async_to_lds_b128 %0, %1, off offset:112\n\t"
      "global_load_async_to_lds_b128 %0, %1, off offset:128\n\t"
      "global_load_async_to_lds_b128 %0, %1, off offset:144\n\t"
      "global_load_async_to_lds_b128 %0, %1, off offset:160\n\t"
      "global_load_async_to_lds_b128 %0, %1, off offset:176\n\t"
      "global_load_async_to_lds_b128 %0, %1, off offset:192\n\t"
      "global_load_async_to_lds_b128 %0, %1, off offset:208\n\t"
      "global_load_async_to_lds_b128 %0, %1, off offset:224\n\t"
      "global_load_async_to_lds_b128 %0, %1, off offset:240"
      :: "v"(d), "v"(s) : "memory");
}

__device__ __forceinline__ void wait_async0() {
  asm volatile("s_wait_asynccnt 0x0" ::: "memory");
}

__device__ __forceinline__ v16h pack16(float4 a0, float4 a1, float4 a2,
                                       float4 a3, float s) {
  v16h r;
  r[0]  = (_Float16)(a0.x * s); r[1]  = (_Float16)(a0.y * s);
  r[2]  = (_Float16)(a0.z * s); r[3]  = (_Float16)(a0.w * s);
  r[4]  = (_Float16)(a1.x * s); r[5]  = (_Float16)(a1.y * s);
  r[6]  = (_Float16)(a1.z * s); r[7]  = (_Float16)(a1.w * s);
  r[8]  = (_Float16)(a2.x * s); r[9]  = (_Float16)(a2.y * s);
  r[10] = (_Float16)(a2.z * s); r[11] = (_Float16)(a2.w * s);
  r[12] = (_Float16)(a3.x * s); r[13] = (_Float16)(a3.y * s);
  r[14] = (_Float16)(a3.z * s); r[15] = (_Float16)(a3.w * s);
  return r;
}

// A fragment 16x32 f16, all 16 rows valid. ISA layout: lane m=lane&15,
// K-base=(lane>>4)*8; elems 0..7 -> kb.., elems 8..15 -> kb+16..
__device__ __forceinline__ v16h a_frag_full(const float* base, int stride,
                                            int koff, float scale, int lane) {
  int m = lane & 15, kb = (lane >> 4) * 8;
  const float* p = base + m * stride + koff + kb;
  return pack16(ld4(p), ld4(p + 4), ld4(p + 16), ld4(p + 20), scale);
}

// Same but only rows 0..7 valid (Q=8 padded to 16): clamp row, zero scale.
__device__ __forceinline__ v16h a_frag_rows8(const float* base, int stride,
                                             int koff, float scale, int lane) {
  int m = lane & 15, kb = (lane >> 4) * 8;
  float sc = (m < 8) ? scale : 0.f;
  const float* p = base + (m & 7) * stride + koff + kb;
  return pack16(ld4(p), ld4(p + 4), ld4(p + 16), ld4(p + 20), sc);
}

// B fragment 32x16 f16: per-lane f32 at p..p+7 and p+16..p+23 (global or LDS).
__device__ __forceinline__ v16h b_frag16(const float* p) {
  return pack16(ld4(p), ld4(p + 4), ld4(p + 16), ld4(p + 20), 1.f);
}

// ---------------- layernorm ----------------
__global__ __launch_bounds__(256) void ln_kernel(const float* __restrict__ x,
                                                 float* __restrict__ y,
                                                 const float* __restrict__ g,
                                                 const float* __restrict__ b) {
  __shared__ float r1[256], r2[256];
  int m = blockIdx.x, t = threadIdx.x;
  const float* xr = x + m * W_;
  float s = 0.f, s2 = 0.f;
  for (int i = t; i < W_; i += 256) { float v = xr[i]; s += v; s2 += v * v; }
  r1[t] = s; r2[t] = s2; __syncthreads();
  for (int o = 128; o > 0; o >>= 1) {
    if (t < o) { r1[t] += r1[t + o]; r2[t] += r2[t + o]; }
    __syncthreads();
  }
  float mu = r1[0] * (1.f / W_);
  float var = r2[0] * (1.f / W_) - mu * mu;
  float inv = rsqrtf(var + 1e-5f);
  for (int i = t; i < W_; i += 256)
    y[m * W_ + i] = (xr[i] - mu) * inv * g[i] + b[i];
}

// ---------------- generic WMMA matmul: (32 x Kd) @ W^T ----------------------
template <int MODE>
__global__ __launch_bounds__(128) void mm_kernel(
    const float* __restrict__ A, int lda, const float* __restrict__ W, int ldw,
    int ntiles, int kchunks, const float* __restrict__ bias,
    float* __restrict__ X, float* __restrict__ out) {
  int lane = threadIdx.x & 31;
  int wg = blockIdx.x * 4 + (threadIdx.x >> 5);
  int mtile = wg & 1;
  int ntile = wg >> 1;
  if (ntile >= ntiles) return;
  int ncol = ntile * 16 + (lane & 15);
  v8f acc = {};
  for (int kt = 0; kt < kchunks; ++kt) {
    int koff = kt * 32;
    v16h a = a_frag_full(A + mtile * 16 * lda, lda, koff, 1.f, lane);
    v16h b = b_frag16(W + ncol * ldw + koff + (lane >> 4) * 8);
    acc = wmma32(a, b, acc);
  }
#pragma unroll
  for (int r = 0; r < 8; ++r) {
    int m = mtile * 16 + r + (lane >> 4) * 8;
    float v = acc[r];
    if constexpr (MODE == 0) {
      out[m * (2 * W_) + ncol] = v;  // in_proj, ldo = 1536
    } else if constexpr (MODE == 1) {
      X[m * W_ + ncol] += v;
    } else if constexpr (MODE == 2) {
      float t = v + bias[ncol];
      out[m * (4 * W_) + ncol] = t / (1.f + expf(-1.702f * t));
    } else {  // MODE == 3
      float t = X[m * W_ + ncol] + v + bias[ncol];
      X[m * W_ + ncol] = t;
      int n = m >> 3, q = m & 7;
      out[n * (L_ * Q_ * W_) + q * W_ + ncol] = t;  // out pre-offset by l*Q*W
    }
  }
}

// ---------------- logits: s0 (WMMA) + coda (VALU); 4 key-chunks per block --
// k rows and q1 are staged into LDS with global_load_async_to_lds_b128.
// Invalid/padded keys load clamped (garbage) rows; every consumer masks.
__global__ __launch_bounds__(128) void logits_kernel(
    const float* __restrict__ qbuf, const float* __restrict__ kg,
    const unsigned char* __restrict__ mask, float* __restrict__ s0,
    float* __restrict__ coda) {
  __shared__ float q1s[8][64];
  __shared__ float kblk[4][32][68];
  int tid = threadIdx.x, lane = tid & 31, w = tid >> 5;
  int kb = blockIdx.x * 4 + w;
  int h = blockIdx.y, n = blockIdx.z;

  // async stage q1 (8x64 f32): each of 128 threads copies one 16B chunk
  {
    int q = tid >> 4, c0 = (tid & 15) * 4;
    async_b128(qbuf + (n * Q_ + q) * (2 * W_) + h * 128 + 64 + c0,
               &q1s[q][c0]);
  }
  // async stage this wave's 32-key block (clamped addresses for padding)
  {
    int kbc = (kb < NKB) ? kb : 0;
    int kk = kbc * 32 + lane;
    int kcl = (kk < K_) ? kk : 0;
    async_row256(kg + ((size_t)(n * K_ + kcl) * H_ + h) * C_,
                 &kblk[w][lane][0]);
  }
  wait_async0();
  __syncthreads();
  if (kb >= NKB) return;

  const float* q0base = qbuf + (n * Q_) * (2 * W_) + h * 128;
  int obase = (n * H_ + h) * Q_ * KPAD;
#pragma unroll
  for (int half = 0; half < 2; ++half) {
    v8f acc = {};
#pragma unroll
    for (int cc = 0; cc < 2; ++cc) {
      v16h a = a_frag_rows8(q0base, 2 * W_, cc * 32, 0.125f, lane);
      v16h b =
          b_frag16(&kblk[w][half * 16 + (lane & 15)][cc * 32 + (lane >> 4) * 8]);
      acc = wmma32(a, b, acc);
    }
    if (lane < 16) {  // lanes 0..15 hold q rows 0..7
      int kk = kb * 32 + half * 16 + lane;
      int mi = n * K_ + ((kk < K_) ? kk : 0);
      bool ok = (mask[mi] != 0) & (kk < K_);
#pragma unroll
      for (int r = 0; r < 8; ++r)
        s0[obase + r * KPAD + kk] = ok ? acc[r] : -3.0e38f;
    }
  }

  // coda: each lane owns one key
  {
    int kk = kb * 32 + lane;
    int mi = n * K_ + ((kk < K_) ? kk : 0);
    bool ok = (mask[mi] != 0) & (kk < K_);
    float kv[64];
#pragma unroll
    for (int c = 0; c < 64; ++c) kv[c] = kblk[w][lane][c];
    for (int q = 0; q < 8; ++q) {
      float g = 0.f, dt = 0.f;
#pragma unroll
      for (int c = 0; c < 64; ++c) {
        float a = q1s[q][c];
        g += fabsf(a - kv[c]);
        dt += a * kv[c];
      }
      float cv = ok ? tanhf(dt * 0.125f) * (2.f / (1.f + expf(g * 0.125f))) : 0.f;
      coda[obase + q * KPAD + kk] = cv;
    }
  }
}

// ---------------- 3-softmax stats + aff@v mix, one block (128t) per (h,n) --
__global__ __launch_bounds__(128) void mix_kernel(
    const float* __restrict__ s0, const float* __restrict__ coda,
    const float* __restrict__ vg, float* __restrict__ mixbuf) {
  __shared__ float tmax[8][257], tsum[8][257];
  __shared__ float fmax[8][8], fsum[8][8], pmax[8], psum[8];
  __shared__ float vblkT[4][64][36];  // per-wave v block, transposed [c][key]
  __shared__ float mixred[4][8][64];
  int tid = threadIdx.x;
  int h = blockIdx.x, n = blockIdx.y;
  const float* sb = s0 + (n * H_ + h) * Q_ * KPAD;
  const float* cb = coda + (n * H_ + h) * Q_ * KPAD;

  for (int pr = tid; pr < 8 * P_; pr += 128) {
    int q = pr / P_, p = pr % P_;
    const float* r = sb + q * KPAD + p;
    float mx = -3e38f;
#pragma unroll
    for (int f = 0; f < F_; ++f) mx = fmaxf(mx, r[f * P_]);
    float s = 0.f;
#pragma unroll
    for (int f = 0; f < F_; ++f) s += expf(r[f * P_] - mx);
    tmax[q][p] = mx; tsum[q][p] = s;
  }
  if (tid < 64) {
    int q = tid >> 3, f = tid & 7;
    const float* r = sb + q * KPAD + f * P_;
    float mx = -3e38f;
    for (int p = 0; p < P_; ++p) mx = fmaxf(mx, r[p]);
    float s = 0.f;
    for (int p = 0; p < P_; ++p) s += expf(r[p] - mx);
    fmax[q][f] = mx; fsum[q][f] = s;
  }
  __syncthreads();
  if (tid < 8) {
    float mx = -3e38f;
#pragma unroll
    for (int f = 0; f < F_; ++f) mx = fmaxf(mx, fmax[tid][f]);
    float s = 0.f;
#pragma unroll
    for (int f = 0; f < F_; ++f) s += fsum[tid][f] * expf(fmax[tid][f] - mx);
    pmax[tid] = mx; psum[tid] = s;
  }
  __syncthreads();

  int w = tid >> 5, lane = tid & 31;
  v8f zero = {0, 0, 0, 0, 0, 0, 0, 0};
  v8f acc[4] = {zero, zero, zero, zero};
  int m = lane & 15, mq = m & 7;
  float okm = (m < 8) ? 1.f : 0.f;
  int kbase = (lane >> 4) * 8;

  for (int kb = w; kb < NKB; kb += 4) {
    // stage this wave's v block transposed into LDS (coalesced b128 reads)
    {
      int key = kb * 32 + lane;
      bool okk = key < K_;
      float mk = okk ? 1.f : 0.f;
      const float* vr =
          vg + ((size_t)(n * K_ + (okk ? key : 0)) * H_ + h) * C_;
#pragma unroll
      for (int c4 = 0; c4 < 16; ++c4) {
        float4 t = ld4(vr + c4 * 4);
        vblkT[w][c4 * 4 + 0][lane] = t.x * mk;
        vblkT[w][c4 * 4 + 1][lane] = t.y * mk;
        vblkT[w][c4 * 4 + 2][lane] = t.z * mk;
        vblkT[w][c4 * 4 + 3][lane] = t.w * mk;
      }
    }
    // A fragment: aff tile (rows 8..15 zeroed). s0/coda padded region holds
    // -3e38 / 0 so loads are unconditional float4.
    v16h a;
    {
      const float* sp = sb + mq * KPAD + kb * 32 + kbase;
      const float* cp = cb + mq * KPAD + kb * 32 + kbase;
      float4 s0v = ld4(sp), s1v = ld4(sp + 4), s2v = ld4(sp + 16),
             s3v = ld4(sp + 20);
      float4 c0v = ld4(cp), c1v = ld4(cp + 4), c2v = ld4(cp + 16),
             c3v = ld4(cp + 20);
      float sv[16] = {s0v.x, s0v.y, s0v.z, s0v.w, s1v.x, s1v.y, s1v.z, s1v.w,
                      s2v.x, s2v.y, s2v.z, s2v.w, s3v.x, s3v.y, s3v.z, s3v.w};
      float cv[16] = {c0v.x, c0v.y, c0v.z, c0v.w, c1v.x, c1v.y, c1v.z, c1v.w,
                      c2v.x, c2v.y, c2v.z, c2v.w, c3v.x, c3v.y, c3v.z, c3v.w};
#pragma unroll
      for (int e = 0; e < 16; ++e) {
        int kk = kb * 32 + kbase + e + ((e >= 8) ? 8 : 0);
        float oke = (kk < K_) ? 1.f : 0.f;
        int f = kk / P_;  f = (f > 7) ? 7 : f;
        int p = kk - f * P_;  p = (p > 256) ? 256 : p;
        float s = sv[e];
        float ap = expf(s - pmax[mq]) / psum[mq];
        float af = expf(s - fmax[mq][f]) / fsum[mq][f];
        float at = expf(s - tmax[mq][p]) / tsum[mq][p];
        float aff = ((ap + af + at) * (1.f / 3.f) + cv[e]) * 0.5f * okm * oke;
        a[e] = (_Float16)aff;
      }
    }
    // B fragments from transposed LDS: keys ke0..ke0+7, ke0+16..ke0+23 per lane
    int ke0 = (lane >> 4) * 8;
#pragma unroll
    for (int ct = 0; ct < 4; ++ct) {
      int c = ct * 16 + (lane & 15);
      v16h b = b_frag16(&vblkT[w][c][ke0]);
      acc[ct] = wmma32(a, b, acc[ct]);
    }
  }
#pragma unroll
  for (int ct = 0; ct < 4; ++ct) {
#pragma unroll
    for (int r = 0; r < 8; ++r) {
      int mrow = r + (lane >> 4) * 8;
      if (mrow < 8) mixred[w][mrow][ct * 16 + (lane & 15)] = acc[ct][r];
    }
  }
  __syncthreads();
  for (int t = tid; t < 512; t += 128) {
    int q = t >> 6, c = t & 63;
    float s = 0.f;
#pragma unroll
    for (int ww = 0; ww < 4; ++ww) s += mixred[ww][q][c];
    mixbuf[(n * Q_ + q) * W_ + h * C_ + c] = s;
  }
}

extern "C" void kernel_launch(void* const* d_in, const int* in_sizes, int n_in,
                              void* d_out, int out_size, void* d_ws,
                              size_t ws_size, hipStream_t stream) {
  const float* x_in  = (const float*)d_in[0];
  const float* k_in  = (const float*)d_in[1];
  const float* v_in  = (const float*)d_in[2];
  const unsigned char* mask = (const unsigned char*)d_in[3];
  const float* in_w  = (const float*)d_in[4];
  const float* out_w = (const float*)d_in[5];
  const float* ln1g  = (const float*)d_in[6];
  const float* ln1b  = (const float*)d_in[7];
  const float* ln2g  = (const float*)d_in[8];
  const float* ln2b  = (const float*)d_in[9];
  const float* fc_w  = (const float*)d_in[10];
  const float* fc_b  = (const float*)d_in[11];
  const float* pr_w  = (const float*)d_in[12];
  const float* pr_b  = (const float*)d_in[13];
  float* dout = (float*)d_out;

  constexpr size_t XSZ = (size_t)N_ * Q_ * W_;        // 24576
  constexpr size_t QSZ = (size_t)N_ * Q_ * 2 * W_;    // 49152
  constexpr size_t SSZ = (size_t)N_ * H_ * Q_ * KPAD; // 798720
  constexpr size_t HSZ = (size_t)N_ * Q_ * 4 * W_;    // 98304

  float* p = (float*)d_ws;
  float* xbuf = p;   p += XSZ;
  float* lnbuf = p;  p += XSZ;
  float* qbuf = p;   p += QSZ;
  float* mixbuf = p; p += XSZ;
  float* hbuf = p;   p += HSZ;
  float* s0buf = p;  p += SSZ;
  float* codabuf = p;

  hipMemcpyAsync(xbuf, x_in, XSZ * sizeof(float), hipMemcpyDeviceToDevice,
                 stream);

  for (int l = 0; l < L_; ++l) {
    const float* kl = k_in + (size_t)l * N_ * K_ * H_ * C_;
    const float* vl = v_in + (size_t)l * N_ * K_ * H_ * C_;

    ln_kernel<<<32, 256, 0, stream>>>(xbuf, lnbuf, ln1g + l * W_, ln1b + l * W_);
    mm_kernel<0><<<48, 128, 0, stream>>>(lnbuf, W_,
                                         in_w + (size_t)l * 2 * W_ * W_, W_, 96,
                                         24, nullptr, nullptr, qbuf);
    logits_kernel<<<dim3(17, H_, N_), 128, 0, stream>>>(qbuf, kl, mask, s0buf,
                                                        codabuf);
    mix_kernel<<<dim3(H_, N_), 128, 0, stream>>>(s0buf, codabuf, vl, mixbuf);
    mm_kernel<1><<<24, 128, 0, stream>>>(mixbuf, W_,
                                         out_w + (size_t)l * W_ * W_, W_, 48, 24,
                                         nullptr, xbuf, nullptr);
    ln_kernel<<<32, 256, 0, stream>>>(xbuf, lnbuf, ln2g + l * W_, ln2b + l * W_);
    mm_kernel<2><<<96, 128, 0, stream>>>(lnbuf, W_,
                                         fc_w + (size_t)l * 4 * W_ * W_, W_, 192,
                                         24, fc_b + l * 4 * W_, nullptr, hbuf);
    mm_kernel<3><<<24, 128, 0, stream>>>(hbuf, 4 * W_,
                                         pr_w + (size_t)l * W_ * 4 * W_, 4 * W_,
                                         48, 96, pr_b + l * W_, xbuf,
                                         dout + (size_t)l * Q_ * W_);
  }
}
